// Encoder_Base_56367150793585
// MI455X (gfx1250) — matrix operand order repaired and checked by reading, not compiled, not run on hardware
//
#include <hip/hip_runtime.h>
#include <hip/hip_bf16.h>

#define EPSV 1e-5f

typedef __attribute__((ext_vector_type(16))) __bf16 v16bf;
typedef __attribute__((ext_vector_type(8)))  __bf16 v8bf;
typedef __attribute__((ext_vector_type(8)))  float  v8f;

union Frag16 { v16bf v; v8bf h[2]; };

// ---------------------------------------------------------------------------
// Utility kernels
// ---------------------------------------------------------------------------
__global__ void k_f32_to_bf16(const float* __restrict__ in,
                              __bf16* __restrict__ out, size_t n) {
    size_t i = (size_t)blockIdx.x * blockDim.x + threadIdx.x;
    if (i < n) out[i] = (__bf16)in[i];
}

__global__ void k_zero_h(float* __restrict__ h, __bf16* __restrict__ hb, size_t n) {
    size_t i = (size_t)blockIdx.x * blockDim.x + threadIdx.x;
    if (i < n) { h[i] = 0.0f; hb[i] = (__bf16)0.0f; }
}

// ---------------------------------------------------------------------------
// bf16 WMMA GEMM: C[M x N] = A[M x K](row-major, lda) * B[K x N](row-major, ldb) [+ bias]
// Block tile 128x64, 256 threads = 8 waves in a 4x2 grid; each wave owns a
// 32x32 output patch = 2 A-frags x 2 B-frags -> 4 v_wmma_f32_16x16x32_bf16
// per K-chunk with 2x fragment reuse. A staged row-major (16B-aligned rows),
// B staged TRANSPOSED so B-fragments are contiguous -> ds_load_b128 only.
// Requires M%128==0, N%64==0, K%32==0 (true for all shapes used here).
// ---------------------------------------------------------------------------
#define TM 128
#define TN 64
#define TK 32

__device__ __forceinline__ v8f wmma_bf16(const Frag16& a, const Frag16& b, v8f c) {
    return __builtin_amdgcn_wmma_f32_16x16x32_bf16(false, a.v, false, b.v,
                                                   (short)0, c, false, false);
}

template <bool HAS_BIAS>
__global__ __launch_bounds__(256) void k_gemm_bf16_wmma(
    const __bf16* __restrict__ A, int lda,
    const __bf16* __restrict__ B, int ldb,
    float* __restrict__ C, int ldc,
    const float* __restrict__ bias, int K)
{
    // row strides: 48 elems (96B) and 40 elems (80B) -- both multiples of 16B
    __shared__ __bf16 As[TM][TK + 16];     // 12 KB
    __shared__ __bf16 BsT[TN][TK + 8];     // 5 KB, transposed: BsT[n][k]

    const int tid  = threadIdx.x;
    const int lane = tid & 31;
    const int wave = tid >> 5;
    const int mBase = blockIdx.y * TM;
    const int nBase = blockIdx.x * TN;
    const int wm = (wave & 3) * 32;   // wave row offset inside block tile
    const int wn = (wave >> 2) * 32;  // wave col offset inside block tile

    v8f acc00 = {}, acc01 = {}, acc10 = {}, acc11 = {};

    // cooperative staging indices
    const int aRow = tid >> 1;           // 0..127
    const int aCol = (tid & 1) * 16;     // 0 or 16
    const int bRow = tid >> 3;           // 0..31  (K index)
    const int bCol = (tid & 7) * 8;      // 0..56  (N index)

    // WMMA fragment lane mapping (ISA 7.12.2)
    const int m0  = wm + (lane & 15);
    const int akh = (lane >> 4) * 8;     // A K-half select (elems)
    const int bn  = lane & 15;
    const int bkh = (lane >> 4) * 8;     // B K-half select (elems)

    for (int k0 = 0; k0 < K; k0 += TK) {
        // stage A tile (two 16B vector copies per thread)
        {
            const __bf16* gp = &A[(size_t)(mBase + aRow) * lda + (k0 + aCol)];
            *(v8bf*)&As[aRow][aCol]     = *(const v8bf*)(gp);
            *(v8bf*)&As[aRow][aCol + 8] = *(const v8bf*)(gp + 8);
        }
        // stage B tile transposed (one 16B vector load, 8 scalar LDS stores)
        {
            v8bf gb = *(const v8bf*)&B[(size_t)(k0 + bRow) * ldb + (nBase + bCol)];
#pragma unroll
            for (int j = 0; j < 8; ++j) BsT[bCol + j][bRow] = gb[j];
        }
        __syncthreads();

        Frag16 a0, a1, b0, b1;
        a0.h[0] = *(const v8bf*)&As[m0][akh];
        a0.h[1] = *(const v8bf*)&As[m0][16 + akh];
        a1.h[0] = *(const v8bf*)&As[m0 + 16][akh];
        a1.h[1] = *(const v8bf*)&As[m0 + 16][16 + akh];
        b0.h[0] = *(const v8bf*)&BsT[wn + bn][bkh];
        b0.h[1] = *(const v8bf*)&BsT[wn + bn][16 + bkh];
        b1.h[0] = *(const v8bf*)&BsT[wn + 16 + bn][bkh];
        b1.h[1] = *(const v8bf*)&BsT[wn + 16 + bn][16 + bkh];

        acc00 = wmma_bf16(a0, b0, acc00);
        acc01 = wmma_bf16(a0, b1, acc01);
        acc10 = wmma_bf16(a1, b0, acc10);
        acc11 = wmma_bf16(a1, b1, acc11);
        __syncthreads();
    }

    // store C (f32 16x16 layout: lane N=lane&15, M-half via lane>=16, 8 VGPRs)
    const int cn = lane & 15;
    const int mh = (lane >> 4) * 8;
    const int col0 = nBase + wn + cn;
    const int col1 = col0 + 16;
    float b0v = 0.0f, b1v = 0.0f;
    if (HAS_BIAS) { b0v = bias[col0]; b1v = bias[col1]; }
#pragma unroll
    for (int i = 0; i < 8; ++i) {
        int rowA = mBase + wm + mh + i;
        int rowB = rowA + 16;
        C[(size_t)rowA * ldc + col0] = acc00[i] + b0v;
        C[(size_t)rowA * ldc + col1] = acc01[i] + b1v;
        C[(size_t)rowB * ldc + col0] = acc10[i] + b0v;
        C[(size_t)rowB * ldc + col1] = acc11[i] + b1v;
    }
}

// ---------------------------------------------------------------------------
// Row LayerNorm (block per row): out = g*((v-mean)/(sqrt(var+EPS)+EPS)) + be
// ---------------------------------------------------------------------------
template <int NITER>
__global__ __launch_bounds__(256) void k_ln_rows(
    const float* __restrict__ in, float* __restrict__ out,
    const float* __restrict__ g, const float* __restrict__ be, int cols)
{
    __shared__ float ssum[256];
    __shared__ float ssq[256];
    const int tid = threadIdx.x;
    const size_t row = blockIdx.x;
    const float* rp = in + row * (size_t)cols;

    float vals[NITER];
    float s = 0.0f, q = 0.0f;
#pragma unroll
    for (int i = 0; i < NITER; ++i) {
        float v = rp[i * 256 + tid];
        vals[i] = v; s += v; q += v * v;
    }
    ssum[tid] = s; ssq[tid] = q;
    __syncthreads();
    for (int off = 128; off > 0; off >>= 1) {
        if (tid < off) { ssum[tid] += ssum[tid + off]; ssq[tid] += ssq[tid + off]; }
        __syncthreads();
    }
    const float mean = ssum[0] / (float)cols;
    const float var  = ssq[0] / (float)cols - mean * mean;
    const float inv  = 1.0f / (sqrtf(var + EPSV) + EPSV);

    float* op = out + row * (size_t)cols;
#pragma unroll
    for (int i = 0; i < NITER; ++i) {
        int c = i * 256 + tid;
        op[c] = g[c] * ((vals[i] - mean) * inv) + be[c];
    }
}

// ---------------------------------------------------------------------------
// Gate kernel (per timestep): LN over 2H of T2=h@U_zr with gammas[1,:2H],
// s = clip(0.2*(s1[:,:2H]+s2)+0.5, 0, 1); z stored; rh = r*h stored as bf16.
// ---------------------------------------------------------------------------
__global__ __launch_bounds__(256) void k_gates(
    const float* __restrict__ T2,     // B x 2H
    const float* __restrict__ S1,     // (B*T) x 3H (already LN'd)
    const float* __restrict__ gammas, // 2 x 3H
    const float* __restrict__ betas,  // 2 x 3H
    const float* __restrict__ h,      // B x H
    float* __restrict__ zbuf,         // B x H
    __bf16* __restrict__ rhbf,        // B x H
    int t_step)
{
    constexpr int H = 1024, T = 128, C = 2 * H, NITER = C / 256;
    __shared__ float ssum[256];
    __shared__ float ssq[256];
    const int tid = threadIdx.x;
    const size_t b = blockIdx.x;
    const float* rp = T2 + b * (size_t)C;

    float vals[NITER];
    float s = 0.0f, q = 0.0f;
#pragma unroll
    for (int i = 0; i < NITER; ++i) {
        float v = rp[i * 256 + tid];
        vals[i] = v; s += v; q += v * v;
    }
    ssum[tid] = s; ssq[tid] = q;
    __syncthreads();
    for (int off = 128; off > 0; off >>= 1) {
        if (tid < off) { ssum[tid] += ssum[tid + off]; ssq[tid] += ssq[tid + off]; }
        __syncthreads();
    }
    const float mean = ssum[0] / (float)C;
    const float var  = ssq[0] / (float)C - mean * mean;
    const float inv  = 1.0f / (sqrtf(var + EPSV) + EPSV);

    const float* s1row = S1 + (b * T + (size_t)t_step) * (3 * H);
#pragma unroll
    for (int i = 0; i < NITER; ++i) {
        int c = i * 256 + tid;
        float s2 = gammas[3 * H + c] * ((vals[i] - mean) * inv) + betas[3 * H + c];
        float sv = 0.2f * (s1row[c] + s2) + 0.5f;
        sv = fminf(fmaxf(sv, 0.0f), 1.0f);
        if (c < H) {
            zbuf[b * H + c] = sv;                   // z gate
        } else {
            int j = c - H;                          // r gate
            float rh = sv * h[b * H + j];
            rhbf[b * H + j] = (__bf16)rh;
        }
    }
}

// ---------------------------------------------------------------------------
// Candidate + state update (per timestep): LN over H of T3=(r*h)@U_c with
// gammas[1,2H:], cand = tanh(s1[:,2H:] + ln); h = z*h + (1-z)*cand; emit out.
// ---------------------------------------------------------------------------
__global__ __launch_bounds__(256) void k_cand_update(
    const float* __restrict__ T3,     // B x H
    const float* __restrict__ S1,     // (B*T) x 3H
    const float* __restrict__ gammas,
    const float* __restrict__ betas,
    const float* __restrict__ zbuf,   // B x H
    float* __restrict__ h,            // B x H (updated in place)
    __bf16* __restrict__ hbf,         // B x H bf16 mirror
    float* __restrict__ out,          // B x T x H
    int t_step)
{
    constexpr int H = 1024, T = 128, NITER = H / 256;
    __shared__ float ssum[256];
    __shared__ float ssq[256];
    const int tid = threadIdx.x;
    const size_t b = blockIdx.x;
    const float* rp = T3 + b * (size_t)H;

    float vals[NITER];
    float s = 0.0f, q = 0.0f;
#pragma unroll
    for (int i = 0; i < NITER; ++i) {
        float v = rp[i * 256 + tid];
        vals[i] = v; s += v; q += v * v;
    }
    ssum[tid] = s; ssq[tid] = q;
    __syncthreads();
    for (int off = 128; off > 0; off >>= 1) {
        if (tid < off) { ssum[tid] += ssum[tid + off]; ssq[tid] += ssq[tid + off]; }
        __syncthreads();
    }
    const float mean = ssum[0] / (float)H;
    const float var  = ssq[0] / (float)H - mean * mean;
    const float inv  = 1.0f / (sqrtf(var + EPSV) + EPSV);

    const float* s1row = S1 + (b * T + (size_t)t_step) * (3 * H) + 2 * H;
#pragma unroll
    for (int i = 0; i < NITER; ++i) {
        int c = i * 256 + tid;
        float lnv  = gammas[3 * H + 2 * H + c] * ((vals[i] - mean) * inv)
                   + betas[3 * H + 2 * H + c];
        float cand = tanhf(s1row[c] + lnv);
        float z    = zbuf[b * H + c];
        float hv   = h[b * H + c];
        float hn   = z * hv + (1.0f - z) * cand;
        h[b * H + c]   = hn;
        hbf[b * H + c] = (__bf16)hn;
        out[(b * T + (size_t)t_step) * H + c] = hn;
    }
}

// ---------------------------------------------------------------------------
// Host launcher
// ---------------------------------------------------------------------------
extern "C" void kernel_launch(void* const* d_in, const int* in_sizes, int n_in,
                              void* d_out, int out_size, void* d_ws, size_t ws_size,
                              hipStream_t stream) {
    (void)in_sizes; (void)n_in; (void)out_size; (void)ws_size;
    const float* x      = (const float*)d_in[0];
    const float* W      = (const float*)d_in[1];
    const float* U      = (const float*)d_in[2];
    const float* bvec   = (const float*)d_in[3];
    const float* gammas = (const float*)d_in[4];
    const float* betas  = (const float*)d_in[5];
    float* out = (float*)d_out;

    constexpr int B = 256, T = 128, H = 1024, INNER = 3 * H;
    const size_t BT = (size_t)B * T;

    // carve scratch from d_ws
    char* ws = (char*)d_ws;
    auto carve = [&](size_t bytes) {
        char* p = ws;
        ws += (bytes + 255) & ~(size_t)255;
        return p;
    };
    __bf16* Xb  = (__bf16*)carve(BT * H * sizeof(__bf16));
    __bf16* Wb  = (__bf16*)carve((size_t)H * INNER * sizeof(__bf16));
    __bf16* Ub  = (__bf16*)carve((size_t)H * INNER * sizeof(__bf16));
    float*  S1  = (float*) carve(BT * INNER * sizeof(float));
    float*  h   = (float*) carve((size_t)B * H * sizeof(float));
    __bf16* hb  = (__bf16*)carve((size_t)B * H * sizeof(__bf16));
    float*  T2  = (float*) carve((size_t)B * 2 * H * sizeof(float));
    float*  T3  = (float*) carve((size_t)B * H * sizeof(float));
    float*  zb  = (float*) carve((size_t)B * H * sizeof(float));
    __bf16* rhb = (__bf16*)carve((size_t)B * H * sizeof(__bf16));

    // precision staging
    {
        size_t n = BT * H;
        k_f32_to_bf16<<<(n + 255) / 256, 256, 0, stream>>>(x, Xb, n);
        n = (size_t)H * INNER;
        k_f32_to_bf16<<<(n + 255) / 256, 256, 0, stream>>>(W, Wb, n);
        k_f32_to_bf16<<<(n + 255) / 256, 256, 0, stream>>>(U, Ub, n);
        n = (size_t)B * H;
        k_zero_h<<<(n + 255) / 256, 256, 0, stream>>>(h, hb, n);
    }

    // Phase 1: S1raw = X @ W + b over all B*T rows (single big WMMA GEMM)
    {
        dim3 grid(INNER / TN, (unsigned)(BT / TM));
        k_gemm_bf16_wmma<true><<<grid, 256, 0, stream>>>(Xb, H, Wb, INNER,
                                                         S1, INNER, bvec, H);
    }
    // Phase 2: row LN over 3072 cols with gammas[0]/betas[0] (in place)
    k_ln_rows<12><<<(unsigned)BT, 256, 0, stream>>>(S1, S1, gammas, betas, INNER);

    // Phase 3: recurrence over T steps
    for (int t = 0; t < T; ++t) {
        // T2 = h @ U[:, :2H]
        dim3 g2((2 * H) / TN, B / TM);
        k_gemm_bf16_wmma<false><<<g2, 256, 0, stream>>>(hb, H, Ub, INNER,
                                                        T2, 2 * H, nullptr, H);
        k_gates<<<B, 256, 0, stream>>>(T2, S1, gammas, betas, h, zb, rhb, t);

        // T3 = (r*h) @ U[:, 2H:]
        dim3 g3(H / TN, B / TM);
        k_gemm_bf16_wmma<false><<<g3, 256, 0, stream>>>(rhb, H, Ub + 2 * H, INNER,
                                                        T3, H, nullptr, H);
        k_cand_update<<<B, 256, 0, stream>>>(T3, S1, gammas, betas, zb, h, hb,
                                             out, t);
    }
}
